// SelfAttn_15281493639295
// MI455X (gfx1250) — matrix operand order repaired and checked by reading, not compiled, not run on hardware
//
#include <hip/hip_runtime.h>
#include <hip/hip_bf16.h>

// ---------------------------------------------------------------------------
// Spatial self-attention (B=32, C=256, H=W=32  ->  N=1024 tokens, dim 256)
// All 5 GEMMs run on v_wmma_f32_16x16x32_bf16; attention is flash-style
// (online softmax, no B*N*N materialization). ~51 GFLOP total.
// ---------------------------------------------------------------------------

#define DEVFN __device__ __forceinline__

typedef __attribute__((ext_vector_type(16))) __bf16 v16bf;
typedef __attribute__((ext_vector_type(8)))  float  v8f;

static constexpr int Bn = 32;    // batch
static constexpr int Cc = 256;   // channels (head dim)
static constexpr int Nn = 1024;  // H*W tokens
static constexpr int KW = 65536; // C*C weight elems

union FragU { v16bf v; uint4 q[2]; };

// round-to-nearest-even fp32 -> bf16
DEVFN unsigned short f2bf(float x) {
    unsigned int u = __builtin_bit_cast(unsigned int, x);
    u += 0x7FFFu + ((u >> 16) & 1u);
    return (unsigned short)(u >> 16);
}

// A fragment of V_WMMA_F32_16X16X32_BF16 from a row-major [M][K] bf16 matrix:
// lane<16: row=lane,    halfwords = K{c0..c0+7, c0+16..c0+23}
// lane>=16: row=lane-16, halfwords = K{c0+8..15, c0+24..31}
DEVFN v16bf load_frag_a(const unsigned short* base, int ld, int row, int c0, int lane) {
    const unsigned short* p = base + (size_t)row * ld + c0 + ((lane & 16) ? 8 : 0);
    FragU f;
    f.q[0] = *(const uint4*)(p);
    f.q[1] = *(const uint4*)(p + 16);
    return f.v;
}

// B fragment (K x N) sourced from row-major [N][K] storage (i.e. B^T rows):
// lane<16: col=lane,    K = c0..c0+7, c0+16..c0+23
// lane>=16: col=lane-16, K = c0+8..c0+15, c0+24..c0+31
DEVFN v16bf load_frag_b(const unsigned short* base, int ld, int nrow, int c0, int lane) {
    const unsigned short* p = base + (size_t)nrow * ld + c0 + ((lane & 16) ? 8 : 0);
    FragU f;
    f.q[0] = *(const uint4*)(p);
    f.q[1] = *(const uint4*)(p + 16);
    return f.v;
}

DEVFN v8f wmma_bf16(v16bf a, v16bf b, v8f c) {
    return __builtin_amdgcn_wmma_f32_16x16x32_bf16(
        false, a, false, b, (short)0, c, false, false);
}

// --------------------------------------------------------------------------
// Kernel 1: x [B][C][N] fp32  ->  xT [B][N][C] bf16   (LDS-tiled transpose)
// --------------------------------------------------------------------------
__global__ void k_transpose_x(const float* __restrict__ x,
                              unsigned short* __restrict__ xT) {
    __shared__ float tile[32][33];
    const int n0 = blockIdx.x * 32;
    const int c0 = blockIdx.y * 32;
    const int b  = blockIdx.z;
    const int tx = threadIdx.x & 31;
    const int ty = threadIdx.x >> 5;            // 0..7
    const float* xb = x + ((size_t)b * Cc + c0) * Nn + n0;
#pragma unroll
    for (int i = 0; i < 4; ++i) {
        int r = ty + i * 8;
        tile[r][tx] = xb[(size_t)r * Nn + tx];  // row=c, col=n (coalesced in n)
    }
    __syncthreads();
    unsigned short* ob = xT + ((size_t)b * Nn + n0) * Cc + c0;
#pragma unroll
    for (int i = 0; i < 4; ++i) {
        int r = ty + i * 8;                     // out row = n0+r, col = c0+tx
        ob[(size_t)r * Cc + tx] = f2bf(tile[tx][r]);
    }
}

// --------------------------------------------------------------------------
// Kernel 2: convert the four [C][C] fp32 weights to bf16 (natural layout)
// --------------------------------------------------------------------------
__global__ void k_conv_w(const float* __restrict__ w0, const float* __restrict__ w1,
                         const float* __restrict__ w2, const float* __restrict__ w3,
                         unsigned short* __restrict__ dst) {
    const int wi = blockIdx.y;
    const float* src = (wi == 0) ? w0 : (wi == 1) ? w1 : (wi == 2) ? w2 : w3;
    const int i = blockIdx.x * 256 + threadIdx.x;
    dst[(size_t)wi * KW + i] = f2bf(src[i]);
}

// --------------------------------------------------------------------------
// Kernel 3: projections.  Per wave: one 16x16 tile, 8 WMMA K-steps (C=256).
//   proj 0/1 : qT/kT [B][N][C]  = xT[n][c] * w[k][c]^T  (scale folded into q)
//   proj 2   : v     [B][C][N]  = wv[k][c] * xT[n][c]^T
// --------------------------------------------------------------------------
__global__ void k_proj(const unsigned short* __restrict__ xT,
                       const unsigned short* __restrict__ wb,   // 4 bf16 weights
                       const float* __restrict__ bq, const float* __restrict__ bk,
                       const float* __restrict__ bv,
                       unsigned short* __restrict__ qT,
                       unsigned short* __restrict__ kT,
                       unsigned short* __restrict__ vv) {
    const int lane  = threadIdx.x & 31;
    const int widx  = threadIdx.x >> 5;
    const int wg    = blockIdx.x * 8 + widx;          // 0 .. 3*B*1024-1
    const int proj  = wg / (Bn * 1024);
    const int rem   = wg % (Bn * 1024);
    const int b     = rem / 1024;
    const int t     = rem % 1024;
    const int col   = lane & 15;
    const int rbase = (lane & 16) ? 8 : 0;

    v8f acc = {};
    if (proj < 2) {
        const int mt = t / 16;                        // n-tile (0..63)
        const int nt = t % 16;                        // kch-tile (0..15)
        const unsigned short* A = xT + (size_t)b * Nn * Cc;
        const unsigned short* Bm = wb + (size_t)proj * KW;   // wq or wk
        const int mrow = mt * 16 + col;
        const int nrow = nt * 16 + col;
#pragma unroll
        for (int s = 0; s < 8; ++s) {
            v16bf af = load_frag_a(A, Cc, mrow, s * 32, lane);
            v16bf bf = load_frag_b(Bm, Cc, nrow, s * 32, lane);
            acc = wmma_bf16(af, bf, acc);
        }
        const float* bias = (proj == 0) ? bq : bk;
        const float bias_v = bias[nt * 16 + col];
        const float sc = (proj == 0) ? 0.0625f : 1.0f;    // 1/sqrt(256)
        unsigned short* outp = (proj == 0) ? qT : kT;
#pragma unroll
        for (int r = 0; r < 8; ++r) {
            int row = rbase + r;                          // token index
            outp[((size_t)b * Nn + mt * 16 + row) * Cc + nt * 16 + col] =
                f2bf((acc[r] + bias_v) * sc);
        }
    } else {
        const int mt = t / 64;                        // kch-tile (0..15)
        const int nt = t % 64;                        // n-tile  (0..63)
        const unsigned short* A = wb + 2 * (size_t)KW;    // wv
        const unsigned short* Bm = xT + (size_t)b * Nn * Cc;
        const int mrow = mt * 16 + col;
        const int nrow = nt * 16 + col;
#pragma unroll
        for (int s = 0; s < 8; ++s) {
            v16bf af = load_frag_a(A, Cc, mrow, s * 32, lane);
            v16bf bf = load_frag_b(Bm, Cc, nrow, s * 32, lane);
            acc = wmma_bf16(af, bf, acc);
        }
#pragma unroll
        for (int r = 0; r < 8; ++r) {
            int kch = mt * 16 + rbase + r;
            vv[((size_t)b * Cc + kch) * Nn + nt * 16 + col] =
                f2bf(acc[r] + bv[kch]);
        }
    }
}

// --------------------------------------------------------------------------
// Kernel 4: flash attention.  One wave owns 16 query rows of one batch.
//   scores: 16 WMMAs per 32-key step; online softmax via shfl_xor + v_exp;
//   D->A fragment transpose through a 1KB per-wave LDS tile; att*V: 16 WMMAs.
// --------------------------------------------------------------------------
__global__ void k_attn(const unsigned short* __restrict__ qT,
                       const unsigned short* __restrict__ kT,
                       const unsigned short* __restrict__ vv,
                       unsigned short* __restrict__ oT) {
    __shared__ unsigned short pls[8][16 * 32];        // per-wave p tile
    const int lane  = threadIdx.x & 31;
    const int widx  = threadIdx.x >> 5;
    const int wg    = blockIdx.x * 8 + widx;          // 0 .. B*64-1
    const int b     = wg / 64;
    const int qt    = wg % 64;
    const int col   = lane & 15;
    const int rbase = (lane & 16) ? 8 : 0;

    const unsigned short* qb = qT + (size_t)b * Nn * Cc;
    const unsigned short* kb = kT + (size_t)b * Nn * Cc;
    const unsigned short* vb = vv + (size_t)b * Cc * Nn;
    unsigned short* myp = pls[widx];

    // resident Q fragments (8 x v16bf = 64 VGPRs), scale already folded in
    v16bf qf[8];
    const int qrow = qt * 16 + col;
#pragma unroll
    for (int s = 0; s < 8; ++s) qf[s] = load_frag_a(qb, Cc, qrow, s * 32, lane);

    float m[8], l[8];
#pragma unroll
    for (int r = 0; r < 8; ++r) { m[r] = -1e30f; l[r] = 0.0f; }
    v8f acc[16];
#pragma unroll
    for (int ct = 0; ct < 16; ++ct) acc[ct] = (v8f){};

    for (int jt = 0; jt < Nn; jt += 32) {
        // ---- scores for keys jt..jt+31 (two 16-col WMMA tiles) ----
        v8f s0 = {}, s1 = {};
        const int krow = jt + col;
        if (jt + 32 < Nn) {                            // hint next K/V tiles
            __builtin_prefetch(kb + (size_t)(krow + 32) * Cc, 0, 1);
            __builtin_prefetch(vb + (size_t)col * Nn + jt + 32, 0, 1);
        }
#pragma unroll
        for (int s = 0; s < 8; ++s) {
            v16bf kf0 = load_frag_b(kb, Cc, krow,      s * 32, lane);
            v16bf kf1 = load_frag_b(kb, Cc, krow + 16, s * 32, lane);
            s0 = wmma_bf16(qf[s], kf0, s0);
            s1 = wmma_bf16(qf[s], kf1, s1);
        }
        // ---- online softmax (row stats live per 16-lane half) ----
        float p0[8], p1[8], fr[8];
#pragma unroll
        for (int r = 0; r < 8; ++r) {
            float a0 = s0[r], a1 = s1[r];
            float mx = fmaxf(a0, a1);
#pragma unroll
            for (int off = 1; off < 16; off <<= 1)
                mx = fmaxf(mx, __shfl_xor(mx, off, 32));
            float mn = fmaxf(m[r], mx);
            p0[r] = __expf(a0 - mn);
            p1[r] = __expf(a1 - mn);
            float rs = p0[r] + p1[r];
#pragma unroll
            for (int off = 1; off < 16; off <<= 1)
                rs += __shfl_xor(rs, off, 32);
            fr[r] = __expf(m[r] - mn);
            l[r]  = l[r] * fr[r] + rs;
            m[r]  = mn;
        }
#pragma unroll
        for (int ct = 0; ct < 16; ++ct)
#pragma unroll
            for (int r = 0; r < 8; ++r) acc[ct][r] *= fr[r];

        // ---- D-layout -> A-fragment transpose through LDS (bf16) ----
#pragma unroll
        for (int r = 0; r < 8; ++r) {
            myp[(rbase + r) * 32 + col]      = f2bf(p0[r]);
            myp[(rbase + r) * 32 + col + 16] = f2bf(p1[r]);
        }
        asm volatile("s_wait_dscnt 0" ::: "memory");   // wave-private tile
        v16bf af = load_frag_a(myp, 32, col, 0, lane);

        // ---- O += P * V  (16 channel tiles) ----
#pragma unroll
        for (int ct = 0; ct < 16; ++ct) {
            v16bf vf = load_frag_b(vb, Nn, ct * 16 + col, jt, lane);
            acc[ct] = wmma_bf16(af, vf, acc[ct]);
        }
    }

    // ---- epilogue: divide by row sum, store oT [B][N][C] bf16 ----
    float inv[8];
#pragma unroll
    for (int r = 0; r < 8; ++r) inv[r] = 1.0f / l[r];
#pragma unroll
    for (int ct = 0; ct < 16; ++ct)
#pragma unroll
        for (int r = 0; r < 8; ++r) {
            int row = qt * 16 + rbase + r;
            oT[((size_t)b * Nn + row) * Cc + ct * 16 + col] =
                f2bf(acc[ct][r] * inv[r]);
        }
}

// --------------------------------------------------------------------------
// Kernel 5: out = wh @ o + bh + x   (fp32 result, [B][C][N])
// --------------------------------------------------------------------------
__global__ void k_final(const unsigned short* __restrict__ oT,
                        const unsigned short* __restrict__ whb,
                        const float* __restrict__ bh,
                        const float* __restrict__ x,
                        float* __restrict__ out) {
    const int lane  = threadIdx.x & 31;
    const int widx  = threadIdx.x >> 5;
    const int wg    = blockIdx.x * 8 + widx;          // 0 .. B*1024-1
    const int b     = wg / 1024;
    const int t     = wg % 1024;
    const int mt    = t / 64;                         // kch-tile
    const int nt    = t % 64;                         // n-tile
    const int col   = lane & 15;
    const int rbase = (lane & 16) ? 8 : 0;

    const unsigned short* Bm = oT + (size_t)b * Nn * Cc;
    const int mrow = mt * 16 + col;
    const int nrow = nt * 16 + col;
    v8f acc = {};
#pragma unroll
    for (int s = 0; s < 8; ++s) {
        v16bf af = load_frag_a(whb, Cc, mrow, s * 32, lane);
        v16bf bf = load_frag_b(Bm,  Cc, nrow, s * 32, lane);
        acc = wmma_bf16(af, bf, acc);
    }
#pragma unroll
    for (int r = 0; r < 8; ++r) {
        int kch = mt * 16 + rbase + r;
        int n   = nt * 16 + col;
        size_t idx = ((size_t)b * Cc + kch) * Nn + n;
        out[idx] = acc[r] + bh[kch] + x[idx];
    }
}

// --------------------------------------------------------------------------
extern "C" void kernel_launch(void* const* d_in, const int* in_sizes, int n_in,
                              void* d_out, int out_size, void* d_ws, size_t ws_size,
                              hipStream_t stream) {
    (void)in_sizes; (void)n_in; (void)out_size; (void)ws_size;
    const float* x  = (const float*)d_in[0];
    const float* wq = (const float*)d_in[1];
    const float* bq = (const float*)d_in[2];
    const float* wk = (const float*)d_in[3];
    const float* bk = (const float*)d_in[4];
    const float* wv = (const float*)d_in[5];
    const float* bv = (const float*)d_in[6];
    const float* wh = (const float*)d_in[7];
    const float* bh = (const float*)d_in[8];
    float* out = (float*)d_out;

    // workspace layout (bf16 elems): xT | 4 weights | qT | kT | v | oT  (~84 MB)
    unsigned short* ws16 = (unsigned short*)d_ws;
    const size_t n_big = (size_t)Bn * Nn * Cc;        // 8,388,608
    unsigned short* xT = ws16;
    unsigned short* wb = xT + n_big;                  // 4 * KW
    unsigned short* qT = wb + 4 * (size_t)KW;
    unsigned short* kT = qT + n_big;
    unsigned short* vv = kT + n_big;
    unsigned short* oT = vv + n_big;

    k_transpose_x<<<dim3(Nn / 32, Cc / 32, Bn), 256, 0, stream>>>(x, xT);
    k_conv_w<<<dim3(KW / 256, 4), 256, 0, stream>>>(wq, wk, wv, wh, wb);
    k_proj<<<(3 * Bn * 1024) / 8, 256, 0, stream>>>(xT, wb, bq, bk, bv, qT, kT, vv);
    k_attn<<<(Bn * 64) / 8, 256, 0, stream>>>(qT, kT, vv, oT);
    k_final<<<(Bn * 1024) / 8, 256, 0, stream>>>(oT, wb + 3 * (size_t)KW, bh, x, out);
}
